// rnns_41283225649561
// MI455X (gfx1250) — compile-verified, matrix operand order checked by reading, not run on hardware
//
#include <hip/hip_runtime.h>
#include <hip/hip_bf16.h>
#include <math.h>

typedef __attribute__((ext_vector_type(16))) _Float16 v16h;
typedef __attribute__((ext_vector_type(8)))  _Float16 v8h;
typedef __attribute__((ext_vector_type(8)))  float    v8f;

#define BB 64
#define TT 128
#define FF 16
#define HH 512
#define G4 2048   // 4*H
#define KT 1024   // 2*H time-embedding width

// ---------------------------------------------------------------------------
// WMMA fragment loader (A or B operand of v_wmma_f32_16x16x32_f16).
// p0 points at element (tile_row0, k0); ld = leading dim in halves.
// Lane L (grp = L/16, r = L%16): holds row r, halves:
//   e in 0..7  -> K = k0 + 8*grp + e
//   e in 8..15 -> K = k0 + 8*grp + e + 8
// i.e. two contiguous 16B chunks at +8*grp and +8*grp+16.
// ---------------------------------------------------------------------------
__device__ __forceinline__ v16h load_frag(const _Float16* p0, int ld) {
  const int lane = threadIdx.x & 31;
  const int grp  = lane >> 4;
  const int r    = lane & 15;
  const _Float16* p = p0 + (size_t)r * ld + 8 * grp;
  v8h lo = *(const v8h*)(p);
  v8h hi = *(const v8h*)(p + 16);
  v16h out;
#pragma unroll
  for (int i = 0; i < 8; ++i) { out[i] = lo[i]; out[i + 8] = hi[i]; }
  return out;
}

__device__ __forceinline__ float sigm(float v) { return 1.0f / (1.0f + expf(-v)); }

// ---------------------------------------------------------------------------
// init: zero step counters, load h0/c0 into state buffers (phase 0)
// ---------------------------------------------------------------------------
__global__ void k_init(const float* __restrict__ h0, const float* __restrict__ c0,
                       float* __restrict__ hbuf, float* __restrict__ cbuf,
                       int* __restrict__ cnt) {
  int idx = blockIdx.x * 256 + threadIdx.x;          // s*32768 + b*512 + n
  int bn  = idx & (BB * HH - 1);
  float hv = h0[bn], cv = c0[bn];
  hbuf[idx] = hv;                                    // phase 0 ([ph][s][b][n])
  cbuf[idx] = cv;                                    // [s][b][n]
  if (idx < 2 * TT) cnt[idx] = 0;
}

// ---------------------------------------------------------------------------
// convert W_ih columns [512:1536) to f16, row-major [2048][1024]
// ---------------------------------------------------------------------------
__global__ void k_cvt_wtime(const float* __restrict__ W_ih, _Float16* __restrict__ Wt) {
  size_t idx = (size_t)blockIdx.x * 256 + threadIdx.x;   // < 2048*1024
  int j = (int)(idx >> 10), k = (int)(idx & 1023);
  Wt[idx] = (_Float16)W_ih[(size_t)j * 1536 + 512 + k];
}

__global__ void k_cvt_whh(const float* __restrict__ W_hh, _Float16* __restrict__ Wh) {
  size_t idx = (size_t)blockIdx.x * 256 + threadIdx.x;   // < 2048*512
  Wh[idx] = (_Float16)W_hh[idx];
}

// ---------------------------------------------------------------------------
// fold segment input linears through W_ih:
//   P[s][k][j]  = sum_m W_is_s[m][k] * W_ih[j][m]           (m < 512)
//   c_all[s][j] = sum_m b_is_s[m] * W_ih[j][m] + b_ih[j] + b_hh[j]
// ---------------------------------------------------------------------------
__global__ void k_proj(const float* __restrict__ W_is0, const float* __restrict__ b_is0,
                       const float* __restrict__ W_is1, const float* __restrict__ b_is1,
                       const float* __restrict__ W_ih,  const float* __restrict__ b_ih,
                       const float* __restrict__ b_hh,
                       float* __restrict__ P, float* __restrict__ c_all) {
  int idx = blockIdx.x * 256 + threadIdx.x;
  if (idx >= 2 * G4) return;
  int s = idx >> 11, j = idx & (G4 - 1);
  const float* Wis = s ? W_is1 : W_is0;
  const float* bis = s ? b_is1 : b_is0;
  int Ks = s ? 6 : 8;
  float accp[8] = {0, 0, 0, 0, 0, 0, 0, 0};
  float accc = 0.0f;
  for (int m = 0; m < HH; ++m) {
    float wih = W_ih[(size_t)j * 1536 + m];
    accc += bis[m] * wih;
    for (int k = 0; k < Ks; ++k) accp[k] += Wis[m * Ks + k] * wih;
  }
#pragma unroll
  for (int k = 0; k < 8; ++k)
    P[((size_t)s * 8 + k) * G4 + j] = (k < Ks) ? accp[k] : 0.0f;
  c_all[s * G4 + j] = accc + b_ih[j] + b_hh[j];
}

// ---------------------------------------------------------------------------
// sinusoidal time embeddings -> f16  Temb[t][b][1024]
// ---------------------------------------------------------------------------
__global__ void k_emb(const float* __restrict__ x, _Float16* __restrict__ Temb) {
  size_t idx = (size_t)blockIdx.x * 256 + threadIdx.x;   // < 8192*1024
  int d = (int)(idx & 1023);
  int row = (int)(idx >> 10);                            // t*64 + b
  int t = row >> 6, b = row & 63;
  int which = d >> 9;
  int dd = d & 511;
  float tval = x[((size_t)b * TT + t) * FF + which];
  int i = dd & 255;
  float freq = expf(-logf(10000.0f) * (float)i * (1.0f / 256.0f));
  float arg = tval * freq;
  float v = (dd < 256) ? cosf(arg) : sinf(arg);
  Temb[idx] = (_Float16)v;
}

// ---------------------------------------------------------------------------
// big GEMM: Gtime[row][j] = sum_k Temb[row][k] * Wt[j][k]  (row = t*64+b)
// f16 WMMA / f32 accumulate, register double-buffered software pipeline:
// loads for k+32 are issued before the 4-WMMA chain for k, so global latency
// overlaps matrix work instead of forcing s_wait_loadcnt 0 per WMMA.
// grid: 2048 blocks x 256 (16384 waves = 512 Mtiles x 32 Ngroups)
// ---------------------------------------------------------------------------
__global__ void k_gemm_time(const _Float16* __restrict__ A, const _Float16* __restrict__ Bw,
                            float* __restrict__ Gtime) {
  int gw = blockIdx.x * 8 + (threadIdx.x >> 5);
  int mt = gw >> 5;                // 0..511
  int ng = gw & 31;                // 0..31
  int row0 = mt * 16;
  int n0 = ng * 64;
  int lane = threadIdx.x & 31, grp = lane >> 4;

  const _Float16* Ap = A + (size_t)row0 * KT;

  v8f acc[4] = {};
  v16h a_cur = load_frag(Ap, KT);
  v16h b_cur[4];
#pragma unroll
  for (int nn = 0; nn < 4; ++nn)
    b_cur[nn] = load_frag(Bw + (size_t)(n0 + nn * 16) * KT, KT);

  for (int k0 = 0; k0 < KT - 32; k0 += 32) {
    // prefetch next k-slice into the shadow buffers
    v16h a_nxt = load_frag(Ap + k0 + 32, KT);
    v16h b_nxt[4];
#pragma unroll
    for (int nn = 0; nn < 4; ++nn)
      b_nxt[nn] = load_frag(Bw + (size_t)(n0 + nn * 16) * KT + k0 + 32, KT);
    // matrix work on current buffers (overlaps the loads above)
#pragma unroll
    for (int nn = 0; nn < 4; ++nn)
      acc[nn] = __builtin_amdgcn_wmma_f32_16x16x32_f16(
          false, a_cur, false, b_cur[nn], (short)0, acc[nn], false, false);
    a_cur = a_nxt;
#pragma unroll
    for (int nn = 0; nn < 4; ++nn) b_cur[nn] = b_nxt[nn];
  }
#pragma unroll
  for (int nn = 0; nn < 4; ++nn)
    acc[nn] = __builtin_amdgcn_wmma_f32_16x16x32_f16(
        false, a_cur, false, b_cur[nn], (short)0, acc[nn], false, false);

#pragma unroll
  for (int nn = 0; nn < 4; ++nn)
#pragma unroll
    for (int vr = 0; vr < 8; ++vr)
      Gtime[(size_t)(row0 + vr + 8 * grp) * G4 + n0 + nn * 16 + (lane & 15)] = acc[nn][vr];
}

// ---------------------------------------------------------------------------
// persistent LSTM recurrence: 32 WGs (2 segs x 16 col-slices), 256 thr (8 waves)
// LDS (192 KB): [0..64K)   h_t tile, f16 [64][512]
//               [64K..192K) this WG's W_hh slice, f16 [4 gates][32 cols][512]
// W_hh slice staged ONCE; per step only h is re-staged. Whole WMMA loop runs
// from LDS (ds_load_b128). Gtime epilogue rows are prefetched during WMMA.
// ---------------------------------------------------------------------------
__global__ void k_recurrent(const float* __restrict__ x, const int* __restrict__ lengths,
                            const _Float16* __restrict__ Whh_h, const float* __restrict__ Gtime,
                            const float* __restrict__ P, const float* __restrict__ c_all,
                            float* __restrict__ hbuf, float* __restrict__ cbuf,
                            float* __restrict__ hfin, int* __restrict__ cnt) {
  extern __shared__ _Float16 smem[];
  _Float16* hs  = smem;                     // 64*512 halves
  _Float16* whh = smem + BB * HH;           // 4*32*512 halves

  const int s     = blockIdx.x >> 4;                // segment
  const int slice = blockIdx.x & 15;                // column slice
  const int n0    = slice * 32;
  const int tid   = threadIdx.x;
  const int wave  = tid >> 5, lane = tid & 31, grp = lane >> 4;
  const int mt    = wave >> 1;                      // batch tile 0..3
  const int ns    = wave & 1;                       // 16-col sub-slice
  const int ncol  = n0 + ns * 16 + (lane & 15);
  const int Ks    = (s == 0) ? 8 : 6;
  const int xoff  = (s == 0) ? 2 : 10;
  const int NWG   = 16;

  // one-time stage of this WG's W_hh slice: whh[g][cl][k] = Whh_h[g*512+n0+cl][k]
  for (int i = tid; i < 4 * 32 * HH; i += 256) {
    int g  = i >> 14;            // /(32*512)
    int rm = i & 16383;
    int cl = rm >> 9;
    int k  = rm & 511;
    whh[i] = Whh_h[(size_t)(g * HH + n0 + cl) * HH + k];
  }
  __syncthreads();

  for (int t = 0; t < TT; ++t) {
    // stage h_t into LDS as f16
    const float* hsrc = hbuf + ((size_t)(t & 1) * 2 + s) * (BB * HH);
    for (int i = tid; i < BB * HH; i += 256) hs[i] = (_Float16)hsrc[i];
    __syncthreads();

    // prefetch this lane's Gtime epilogue rows (hidden behind WMMA loop)
    const float* gt = Gtime + (size_t)t * BB * G4;
#pragma unroll
    for (int vr = 0; vr < 8; ++vr) {
      int b = mt * 16 + vr + 8 * grp;
      __builtin_prefetch(gt + (size_t)b * G4 + ncol, 0, 0);
    }

    v8f acc[4] = {};
    for (int k0 = 0; k0 < HH; k0 += 32) {
      v16h a = load_frag(hs + (size_t)(mt * 16) * HH + k0, HH);
      v16h bf[4];
#pragma unroll
      for (int g = 0; g < 4; ++g)
        bf[g] = load_frag(whh + (size_t)(g * 32 + ns * 16) * HH + k0, HH);
#pragma unroll
      for (int g = 0; g < 4; ++g)
        acc[g] = __builtin_amdgcn_wmma_f32_16x16x32_f16(
            false, a, false, bf[g], (short)0, acc[g], false, false);
    }

#pragma unroll
    for (int vr = 0; vr < 8; ++vr) {
      int b = mt * 16 + vr + 8 * grp;
      float xv[8];
      for (int k = 0; k < Ks; ++k) xv[k] = x[((size_t)b * TT + t) * FF + xoff + k];
      float g4[4];
#pragma unroll
      for (int g = 0; g < 4; ++g) {
        int j = g * HH + ncol;
        float v = acc[g][vr] + gt[(size_t)b * G4 + j] + c_all[s * G4 + j];
        for (int k = 0; k < Ks; ++k) v += xv[k] * P[((size_t)s * 8 + k) * G4 + j];
        g4[g] = v;
      }
      size_t hc = ((size_t)s * BB + b) * HH + ncol;
      float cold = cbuf[hc];
      float cn = sigm(g4[1]) * cold + sigm(g4[0]) * tanhf(g4[2]);
      float hn = sigm(g4[3]) * tanhf(cn);
      cbuf[hc] = cn;
      hbuf[((size_t)((t + 1) & 1) * 2 + s) * (BB * HH) + (size_t)b * HH + ncol] = hn;
      if (lengths[b] - 1 == t) hfin[((size_t)b * 2 + s) * HH + ncol] = hn;
    }

    // device-scope step barrier within this segment
    __threadfence();
    __syncthreads();
    if (tid == 0) {
      int* c = &cnt[s * TT + t];
      __hip_atomic_fetch_add(c, 1, __ATOMIC_ACQ_REL, __HIP_MEMORY_SCOPE_AGENT);
      while (__hip_atomic_load(c, __ATOMIC_ACQUIRE, __HIP_MEMORY_SCOPE_AGENT) < NWG)
        __builtin_amdgcn_s_sleep(1);
    }
    __syncthreads();
  }
}

// ---------------------------------------------------------------------------
// head: out[b][o] = sigmoid(hfin[b] . W_o[o] + b_o[o]),  o < 14
// ---------------------------------------------------------------------------
__global__ void k_out(const float* __restrict__ hfin, const float* __restrict__ W_o,
                      const float* __restrict__ b_o, float* __restrict__ out) {
  int idx = blockIdx.x * 256 + threadIdx.x;
  if (idx >= BB * (FF - 2)) return;
  int b = idx / (FF - 2), o = idx % (FF - 2);
  float acc = b_o[o];
  for (int j = 0; j < 2 * HH; ++j)
    acc += hfin[(size_t)b * 2 * HH + j] * W_o[(size_t)o * 2 * HH + j];
  out[idx] = sigm(acc);
}

// ---------------------------------------------------------------------------
extern "C" void kernel_launch(void* const* d_in, const int* in_sizes, int n_in,
                              void* d_out, int out_size, void* d_ws, size_t ws_size,
                              hipStream_t stream) {
  const float* x      = (const float*)d_in[0];
  const int*   lens   = (const int*)  d_in[1];
  const float* h0     = (const float*)d_in[2];
  const float* c0     = (const float*)d_in[3];
  const float* W_is0  = (const float*)d_in[4];
  const float* b_is0  = (const float*)d_in[5];
  const float* W_is1  = (const float*)d_in[6];
  const float* b_is1  = (const float*)d_in[7];
  const float* W_ih   = (const float*)d_in[8];
  const float* W_hh   = (const float*)d_in[9];
  const float* b_ih   = (const float*)d_in[10];
  const float* b_hh   = (const float*)d_in[11];
  const float* W_o    = (const float*)d_in[12];
  const float* b_o    = (const float*)d_in[13];
  float* out = (float*)d_out;

  char* ws = (char*)d_ws;
  auto alloc = [&](size_t bytes) -> void* {
    void* p = (void*)ws;
    ws += (bytes + 255) & ~(size_t)255;
    return p;
  };
  _Float16* Wtime_h = (_Float16*)alloc((size_t)G4 * KT * sizeof(_Float16)); // 4 MiB
  _Float16* Whh_h   = (_Float16*)alloc((size_t)G4 * HH * sizeof(_Float16)); // 2 MiB
  _Float16* Temb_h  = (_Float16*)alloc((size_t)TT * BB * KT * sizeof(_Float16)); // 16 MiB
  float*    P       = (float*)alloc((size_t)2 * 8 * G4 * sizeof(float));
  float*    c_all   = (float*)alloc((size_t)2 * G4 * sizeof(float));
  float*    Gtime   = (float*)alloc((size_t)TT * BB * G4 * sizeof(float));  // 64 MiB
  float*    hbuf    = (float*)alloc((size_t)2 * 2 * BB * HH * sizeof(float));
  float*    cbuf    = (float*)alloc((size_t)2 * BB * HH * sizeof(float));
  float*    hfin    = (float*)alloc((size_t)BB * 2 * HH * sizeof(float));
  int*      cnt     = (int*)alloc((size_t)2 * TT * sizeof(int));
  (void)ws_size; (void)in_sizes; (void)n_in; (void)out_size;

  k_init<<<256, 256, 0, stream>>>(h0, c0, hbuf, cbuf, cnt);
  k_cvt_wtime<<<8192, 256, 0, stream>>>(W_ih, Wtime_h);
  k_cvt_whh<<<4096, 256, 0, stream>>>(W_hh, Whh_h);
  k_proj<<<16, 256, 0, stream>>>(W_is0, b_is0, W_is1, b_is1, W_ih, b_ih, b_hh, P, c_all);
  k_emb<<<32768, 256, 0, stream>>>(x, Temb_h);
  k_gemm_time<<<2048, 256, 0, stream>>>(Temb_h, Wtime_h, Gtime);
  size_t lds_bytes = (size_t)(BB * HH + 4 * 32 * HH) * sizeof(_Float16);   // 192 KiB
  k_recurrent<<<32, 256, lds_bytes, stream>>>(
      x, lens, Whh_h, Gtime, P, c_all, hbuf, cbuf, hfin, cnt);
  k_out<<<4, 256, 0, stream>>>(hfin, W_o, b_o, out);
}
